// FFTStackedHB_48808008351942
// MI455X (gfx1250) — compile-verified, matrix operand order checked by reading, not run on hardware
//
#include <hip/hip_runtime.h>
#include <hip/hip_bf16.h>
#include <math.h>

// ---------------------------------------------------------------------------
// Types for CDNA5 WMMA / TDM
// ---------------------------------------------------------------------------
typedef __attribute__((ext_vector_type(16))) __bf16 v16bf;
typedef __attribute__((ext_vector_type(8)))  __bf16 v8bf;
typedef __attribute__((ext_vector_type(2)))  __bf16 v2bf;
typedef __attribute__((ext_vector_type(8)))  float  v8f;
typedef __attribute__((ext_vector_type(4)))  unsigned int v4u;
typedef __attribute__((ext_vector_type(8)))  int    v8i;
typedef __attribute__((ext_vector_type(4)))  int    v4i;

#if defined(__has_builtin)
#if __has_builtin(__builtin_amdgcn_tensor_load_to_lds) && \
    __has_builtin(__builtin_amdgcn_s_wait_tensorcnt)
#define HAVE_TDM 1
#warning "CDNA5 probe: TDM builtins available - tensor_load_to_lds path enabled"
#else
#warning "CDNA5 probe: TDM builtins NOT available - cooperative-load fallback"
#endif
#endif

// ---------------------------------------------------------------------------
// Problem constants (match setup_inputs)
// ---------------------------------------------------------------------------
constexpr int BATCH = 8192;
constexpr int D_IN  = 3072;
constexpr int D_H   = 4096;
constexpr int D_OUT = 1000;

constexpr float CCURV  = 0.001f;
constexpr float L_EXPF = 1.0e8f;
constexpr float EPSF   = 1e-7f;
constexpr float BN_EPS = 1e-5f;

// GEMM tiling: block tile 128x128, K-step 32, 8 waves (wave32) in 4x2 grid,
// each wave owns a 32x64 patch = 2x4 WMMA tiles.
constexpr int TM = 128;
constexpr int TN = 128;
constexpr int TK = 32;
// bf16 LDS pitch: 40 elems = 80B. Rows stay 16B-aligned (b128 frag loads) and
// 16 consecutive rows land on 16 distinct 16B-granules across the 64 banks.
constexpr int PITCH = 40;

// ---------------------------------------------------------------------------
// WMMA helpers
// ---------------------------------------------------------------------------
__device__ inline v8f wmma_bf16(v16bf a, v16bf b, v8f c) {
  return __builtin_amdgcn_wmma_f32_16x16x32_bf16(false, a, false, b, (short)0,
                                                 c, false, false);
}

// Fragment halves: elems 0..7 = K kbase..kbase+7, elems 8..15 = kbase+16..+23.
__device__ inline v16bf frag16(const __bf16* base) {
  v8bf f0 = *(const v8bf*)(base);
  v8bf f1 = *(const v8bf*)(base + 16);
  return __builtin_shufflevector(f0, f1, 0, 1, 2, 3, 4, 5, 6, 7, 8, 9, 10, 11,
                                 12, 13, 14, 15);
}

// One K=32 step on a 32x64 wave patch from pre-split bf16 LDS tiles.
// A is [row][k] (pitch 40), B is stored transposed as [col][k] (pitch 40),
// so every fragment half is one ds_load_b128. 24 WMMAs per call.
__device__ inline void wave_mma_step(const __bf16* __restrict__ Ah,
                                     const __bf16* __restrict__ Al,
                                     const __bf16* __restrict__ Bh,
                                     const __bf16* __restrict__ Bl,
                                     int wave_m, int wave_n, int lane,
                                     v8f acc[2][4]) {
  const int mlo = lane & 15;
  const int kb2 = (lane >> 4) << 3;  // 0 or 8

  v16bf ah[2], al[2];
#pragma unroll
  for (int t = 0; t < 2; ++t) {
    const int row = wave_m * 32 + t * 16 + mlo;
    ah[t] = frag16(Ah + row * PITCH + kb2);
    al[t] = frag16(Al + row * PITCH + kb2);
  }
#pragma unroll
  for (int tn = 0; tn < 4; ++tn) {
    const int col = wave_n * 64 + tn * 16 + mlo;
    const v16bf bh = frag16(Bh + col * PITCH + kb2);
    const v16bf bl = frag16(Bl + col * PITCH + kb2);
#pragma unroll
    for (int tm = 0; tm < 2; ++tm) {
      acc[tm][tn] = wmma_bf16(ah[tm], bh, acc[tm][tn]);
      acc[tm][tn] = wmma_bf16(ah[tm], bl, acc[tm][tn]);
      acc[tm][tn] = wmma_bf16(al[tm], bh, acc[tm][tn]);
    }
  }
}

// Split-and-store staging: regs -> (hi,lo) bf16 LDS tiles. A row-major;
// B transposed ([col][k]) with K-pairs so both A and B stage as b32 DS stores.
__device__ inline void stage_tiles(const float2 aR[8], const float2 bR[8],
                                   __bf16* Ah, __bf16* Al, __bf16* Bh,
                                   __bf16* Bl, int tid) {
#pragma unroll
  for (int u = 0; u < 8; ++u) {
    const int p = tid + u * 256;           // 2048 float2 = 128x32 A tile
    const int r = p >> 4, c2 = (p & 15) << 1;
    const float x0 = aR[u].x, x1 = aR[u].y;
    const __bf16 h0 = (__bf16)x0, h1 = (__bf16)x1;
    v2bf th; th[0] = h0; th[1] = h1;
    v2bf tl; tl[0] = (__bf16)(x0 - (float)h0); tl[1] = (__bf16)(x1 - (float)h1);
    *(v2bf*)&Ah[r * PITCH + c2] = th;
    *(v2bf*)&Al[r * PITCH + c2] = tl;
  }
#pragma unroll
  for (int u = 0; u < 8; ++u) {
    const int p  = tid + u * 256;          // 2048 K-pairs = 32x128 B tile
    const int k2 = (p >> 7) << 1, col = p & 127;
    const float w0 = bR[u].x, w1 = bR[u].y;
    const __bf16 h0 = (__bf16)w0, h1 = (__bf16)w1;
    v2bf th; th[0] = h0; th[1] = h1;
    v2bf tl; tl[0] = (__bf16)(w0 - (float)h0); tl[1] = (__bf16)(w1 - (float)h1);
    *(v2bf*)&Bh[col * PITCH + k2] = th;
    *(v2bf*)&Bl[col * PITCH + k2] = tl;
  }
}

// ---------------------------------------------------------------------------
// GEMM1: Z = X @ W1 + b1     [8192,3072] @ [3072,4096]
// ---------------------------------------------------------------------------
__global__ __launch_bounds__(256) void gemm1_kernel(
    const float* __restrict__ X, const float* __restrict__ W1,
    const float* __restrict__ b1, float* __restrict__ Z) {
  __shared__ alignas(16) __bf16 Ah[TM * PITCH];
  __shared__ alignas(16) __bf16 Al[TM * PITCH];
  __shared__ alignas(16) __bf16 Bh[TN * PITCH];
  __shared__ alignas(16) __bf16 Bl[TN * PITCH];

  const int tid    = threadIdx.x;
  const int lane   = tid & 31;
  const int wid    = tid >> 5;
  const int wave_m = wid & 3;
  const int wave_n = wid >> 2;
  const int m0     = blockIdx.y * TM;
  const int n0     = blockIdx.x * TN;

  const v8f zf = {0.f, 0.f, 0.f, 0.f, 0.f, 0.f, 0.f, 0.f};
  v8f acc[2][4];
#pragma unroll
  for (int a = 0; a < 2; ++a)
#pragma unroll
    for (int b = 0; b < 4; ++b) acc[a][b] = zf;

  float2 aR[8], bR[8];
#pragma unroll
  for (int u = 0; u < 8; ++u) {
    const int p = tid + u * 256;
    const int r = p >> 4, c2 = (p & 15) << 1;
    aR[u] = *(const float2*)&X[(size_t)(m0 + r) * D_IN + c2];
  }
#pragma unroll
  for (int u = 0; u < 8; ++u) {
    const int p  = tid + u * 256;
    const int k2 = (p >> 7) << 1, col = n0 + (p & 127);
    bR[u].x = W1[(size_t)k2 * D_H + col];
    bR[u].y = W1[(size_t)(k2 + 1) * D_H + col];
  }

  for (int kb = 0; kb < D_IN; kb += TK) {
    stage_tiles(aR, bR, Ah, Al, Bh, Bl, tid);
    __syncthreads();

    const int kn = kb + TK;
    if (kn < D_IN) {  // issue next-tile loads; they overlap the WMMA burst
      __builtin_prefetch(&X[(size_t)(m0 + (tid >> 1)) * D_IN + kn], 0, 0);
#pragma unroll
      for (int u = 0; u < 8; ++u) {
        const int p = tid + u * 256;
        const int r = p >> 4, c2 = (p & 15) << 1;
        aR[u] = *(const float2*)&X[(size_t)(m0 + r) * D_IN + kn + c2];
      }
#pragma unroll
      for (int u = 0; u < 8; ++u) {
        const int p  = tid + u * 256;
        const int k2 = kn + ((p >> 7) << 1), col = n0 + (p & 127);
        bR[u].x = W1[(size_t)k2 * D_H + col];
        bR[u].y = W1[(size_t)(k2 + 1) * D_H + col];
      }
    }

    wave_mma_step(Ah, Al, Bh, Bl, wave_m, wave_n, lane, acc);
    __syncthreads();
  }

#pragma unroll
  for (int tm = 0; tm < 2; ++tm) {
#pragma unroll
    for (int tn = 0; tn < 4; ++tn) {
      const int row0 = m0 + wave_m * 32 + tm * 16 + ((lane >> 4) << 3);
      const int col  = n0 + wave_n * 64 + tn * 16 + (lane & 15);
      const float bias = b1[col];
#pragma unroll
      for (int i = 0; i < 8; ++i)
        Z[(size_t)(row0 + i) * D_H + col] = acc[tm][tn][i] + bias;
    }
  }
}

// ---------------------------------------------------------------------------
// GEMM2: Out = relu(BN2(H2)) @ W2 + b2   [8192,4096] @ [4096,1000]
// BN2+ReLU fused into the A-tile register load (saves a 134MB round trip).
// ---------------------------------------------------------------------------
__global__ __launch_bounds__(256) void gemm2_kernel(
    const float* __restrict__ H2, const float* __restrict__ W2,
    const float* __restrict__ b2, const float* __restrict__ gamma2,
    const float* __restrict__ beta2, const float* __restrict__ mu2,
    const float* __restrict__ rsig2, float* __restrict__ Out) {
  __shared__ alignas(16) __bf16 Ah[TM * PITCH];
  __shared__ alignas(16) __bf16 Al[TM * PITCH];
  __shared__ alignas(16) __bf16 Bh[TN * PITCH];
  __shared__ alignas(16) __bf16 Bl[TN * PITCH];

  const int tid    = threadIdx.x;
  const int lane   = tid & 31;
  const int wid    = tid >> 5;
  const int wave_m = wid & 3;
  const int wave_n = wid >> 2;
  const int m0     = blockIdx.y * TM;
  const int n0     = blockIdx.x * TN;

  const v8f zf = {0.f, 0.f, 0.f, 0.f, 0.f, 0.f, 0.f, 0.f};
  v8f acc[2][4];
#pragma unroll
  for (int a = 0; a < 2; ++a)
#pragma unroll
    for (int b = 0; b < 4; ++b) acc[a][b] = zf;

  float2 aR[8], bR[8];
#pragma unroll
  for (int u = 0; u < 8; ++u) {
    const int p = tid + u * 256;
    const int r = p >> 4, c2 = (p & 15) << 1;
    float2 t = *(const float2*)&H2[(size_t)(m0 + r) * D_H + c2];
    t.x = fmaxf(gamma2[c2] * (t.x - mu2[c2]) * rsig2[c2] + beta2[c2], 0.f);
    t.y = fmaxf(gamma2[c2 + 1] * (t.y - mu2[c2 + 1]) * rsig2[c2 + 1] +
                beta2[c2 + 1], 0.f);
    aR[u] = t;
  }
#pragma unroll
  for (int u = 0; u < 8; ++u) {
    const int p  = tid + u * 256;
    const int k2 = (p >> 7) << 1, col = n0 + (p & 127);
    const bool ok = col < D_OUT;
    bR[u].x = ok ? W2[(size_t)k2 * D_OUT + col] : 0.f;
    bR[u].y = ok ? W2[(size_t)(k2 + 1) * D_OUT + col] : 0.f;
  }

  for (int kb = 0; kb < D_H; kb += TK) {
    stage_tiles(aR, bR, Ah, Al, Bh, Bl, tid);
    __syncthreads();

    const int kn = kb + TK;
    if (kn < D_H) {
      __builtin_prefetch(&H2[(size_t)(m0 + (tid >> 1)) * D_H + kn], 0, 0);
#pragma unroll
      for (int u = 0; u < 8; ++u) {
        const int p = tid + u * 256;
        const int r = p >> 4, c2 = (p & 15) << 1;
        const int c = kn + c2;
        float2 t = *(const float2*)&H2[(size_t)(m0 + r) * D_H + c];
        t.x = fmaxf(gamma2[c] * (t.x - mu2[c]) * rsig2[c] + beta2[c], 0.f);
        t.y = fmaxf(gamma2[c + 1] * (t.y - mu2[c + 1]) * rsig2[c + 1] +
                    beta2[c + 1], 0.f);
        aR[u] = t;
      }
#pragma unroll
      for (int u = 0; u < 8; ++u) {
        const int p  = tid + u * 256;
        const int k2 = kn + ((p >> 7) << 1), col = n0 + (p & 127);
        const bool ok = col < D_OUT;
        bR[u].x = ok ? W2[(size_t)k2 * D_OUT + col] : 0.f;
        bR[u].y = ok ? W2[(size_t)(k2 + 1) * D_OUT + col] : 0.f;
      }
    }

    wave_mma_step(Ah, Al, Bh, Bl, wave_m, wave_n, lane, acc);
    __syncthreads();
  }

#pragma unroll
  for (int tm = 0; tm < 2; ++tm) {
#pragma unroll
    for (int tn = 0; tn < 4; ++tn) {
      const int row0 = m0 + wave_m * 32 + tm * 16 + ((lane >> 4) << 3);
      const int col  = n0 + wave_n * 64 + tn * 16 + (lane & 15);
      if (col < D_OUT) {
        const float bias = b2[col];
#pragma unroll
        for (int i = 0; i < 8; ++i)
          Out[(size_t)(row0 + i) * D_OUT + col] = acc[tm][tn][i] + bias;
      }
    }
  }
}

// ---------------------------------------------------------------------------
// H_eff[k] = exp(L * log(g_real + i g_imag))  (column filter, precomputed)
// ---------------------------------------------------------------------------
__global__ void heff_kernel(const float* __restrict__ g_real,
                            const float* __restrict__ g_imag,
                            float* __restrict__ hre, float* __restrict__ him) {
  const int k = blockIdx.x * 256 + threadIdx.x;
  const float gr = g_real[k], gi = g_imag[k];
  const float logmag = 0.5f * logf(gr * gr + gi * gi);
  const float ang    = atan2f(gi, gr);
  const float emag   = expf(L_EXPF * logmag);
  float sp, cp;
  __sincosf(L_EXPF * ang, &sp, &cp);
  hre[k] = emag * cp;
  him[k] = emag * sp;
}

// ---------------------------------------------------------------------------
// Deterministic batchnorm stats: 32 row-strips of partials, fixed-order sum.
// ---------------------------------------------------------------------------
__global__ void bn_partial_kernel(const float* __restrict__ Z,
                                  float* __restrict__ part_s,
                                  float* __restrict__ part_s2) {
  const int c  = blockIdx.x * 256 + threadIdx.x;
  const int r0 = blockIdx.y * (BATCH / 32);
  float s = 0.f, s2 = 0.f;
  for (int r = r0; r < r0 + (BATCH / 32); ++r) {
    const float v = Z[(size_t)r * D_H + c];
    s += v;
    s2 += v * v;
  }
  part_s [(size_t)blockIdx.y * D_H + c] = s;
  part_s2[(size_t)blockIdx.y * D_H + c] = s2;
}

__global__ void bn_final_kernel(const float* __restrict__ part_s,
                                const float* __restrict__ part_s2,
                                float* __restrict__ mu,
                                float* __restrict__ rsig) {
  const int c = blockIdx.x * 256 + threadIdx.x;
  float s = 0.f, s2 = 0.f;
#pragma unroll
  for (int i = 0; i < 32; ++i) {
    s  += part_s [(size_t)i * D_H + c];
    s2 += part_s2[(size_t)i * D_H + c];
  }
  const float inv = 1.f / (float)BATCH;
  const float m   = s * inv;
  const float var = s2 * inv - m * m;
  mu[c]   = m;
  rsig[c] = rsqrtf(var + BN_EPS);
}

// ---------------------------------------------------------------------------
// Row kernel helpers
// ---------------------------------------------------------------------------
__device__ inline float block_sum(float v, float* red) {
#pragma unroll
  for (int o = 16; o > 0; o >>= 1) v += __shfl_down(v, o, 32);
  __syncthreads();  // protect red across successive reductions
  if ((threadIdx.x & 31) == 0) red[threadIdx.x >> 5] = v;
  __syncthreads();
  float s = 0.f;
#pragma unroll
  for (int i = 0; i < 8; ++i) s += red[i];
  return s;
}

__device__ inline void fft_stages(float* re, float* im, float sign, int tid) {
  int ls = 0;
  for (int len = 2; len <= D_H; len <<= 1) {
    const int   half = len >> 1;
    const float ang0 = sign * 6.28318530717958647692f / (float)len;
    for (int t = tid; t < (D_H >> 1); t += 256) {
      const int j  = t & (half - 1);
      const int g  = t >> ls;
      const int i0 = g * len + j;
      const int i1 = i0 + half;
      float sw, cw;
      __sincosf(ang0 * (float)j, &sw, &cw);  // TRANS ops co-execute with VALU
      const float ur = re[i0], ui = im[i0];
      const float vr = re[i1], vi = im[i1];
      const float tr = vr * cw - vi * sw;
      const float ti = vr * sw + vi * cw;
      re[i0] = ur + tr;  im[i0] = ui + ti;
      re[i1] = ur - tr;  im[i1] = ui - ti;
    }
    __syncthreads();
    ++ls;
  }
}

#ifdef HAVE_TDM
// TDM 2-D descriptor: one row of `nelem` fp32, global -> LDS (ISA §8.3/8.4).
__device__ inline void tdm_load_row_to_lds(const float* gsrc, unsigned lds_off,
                                           int nelem) {
  const unsigned long long ga = (unsigned long long)(uintptr_t)gsrc;
  v4u g0;
  g0[0] = 1u;                                   // count=1 user descriptor
  g0[1] = lds_off;                              // lds_addr (bytes)
  g0[2] = (unsigned)(ga & 0xffffffffu);         // global_addr[31:0]
  g0[3] = (unsigned)((ga >> 32) & 0x01ffffffu)  // global_addr[56:32]
          | (2u << 30);                         // type=2 ("image")
  v8i g1;
  g1[0] = (int)(2u << 16);                      // data_size=4B, no multicast
  g1[1] = (int)((unsigned)nelem << 16);         // tensor_dim0[15:0]
  g1[2] = (int)(((unsigned)nelem >> 16) | (1u << 16));  // dim0 hi | tensor_dim1=1
  g1[3] = (int)((unsigned)nelem << 16);         // tile_dim0
  g1[4] = 1;                                    // tile_dim1=1, tile_dim2=0
  g1[5] = nelem;                                // tensor_dim0_stride lo
  g1[6] = 0;
  g1[7] = 0;
  const v4i gz = {0, 0, 0, 0};
#if __clang_major__ >= 23
  v8i gz8 = {0, 0, 0, 0, 0, 0, 0, 0};
  __builtin_amdgcn_tensor_load_to_lds(g0, g1, gz, gz, gz8, 0);
#else
  __builtin_amdgcn_tensor_load_to_lds(g0, g1, gz, gz, 0);
#endif
}
#endif

// ---------------------------------------------------------------------------
// Per-row kernel: TDM row->LDS, BN1+ReLU -> FFT -> *H_eff*artanh_scale*(1/N)
// -> iFFT -> exp map -> alpha*y + beta_p*h, in place on Z.
// ---------------------------------------------------------------------------
__global__ __launch_bounds__(256) void row_kernel(
    float* __restrict__ Z, const float* __restrict__ gamma1,
    const float* __restrict__ beta1, const float* __restrict__ mu1,
    const float* __restrict__ rsig1, const float* __restrict__ heff_re,
    const float* __restrict__ heff_im, const float* __restrict__ alpha_p,
    const float* __restrict__ betap_p) {
  __shared__ float sre[D_H];
  __shared__ float sim[D_H];
  __shared__ float zrow[D_H];
  __shared__ float red[8];

  const int    tid = threadIdx.x;
  const size_t row = blockIdx.x;
  float* zr = Z + row * D_H;

#ifdef HAVE_TDM
  if (tid < 32) {  // one wave drives the DMA; TDM ignores EXEC anyway
    tdm_load_row_to_lds(zr, (unsigned)(uintptr_t)&zrow[0], D_H);
    __builtin_amdgcn_s_wait_tensorcnt(0);
  }
  __syncthreads();
#else
  for (int c = tid; c < D_H; c += 256) zrow[c] = zr[c];
  __syncthreads();
#endif

  const float sc = sqrtf(CCURV);

  // BN1 + ReLU, deposit bit-reversed, accumulate ||h||^2
  float ss = 0.f;
  for (int c = tid; c < D_H; c += 256) {
    const float zv = zrow[c];
    const float h  = fmaxf(gamma1[c] * (zv - mu1[c]) * rsig1[c] + beta1[c], 0.f);
    ss += h * h;
    const int j = __brev((unsigned)c) >> 20;  // 12-bit reversal
    sre[j] = h;
    sim[j] = 0.f;
  }
  const float nh    = sqrtf(block_sum(ss, red));
  const float scn   = sc * nh;
  const float scn_c = fminf(fmaxf(scn, EPSF), 1.f - 1e-5f);
  const float u_scale = atanhf(scn_c) / fmaxf(scn, EPSF);

  fft_stages(sre, sim, -1.f, tid);  // forward FFT (natural-order output)

  // Spectral filter; fold u_scale (log map) and 1/N (ifft) — both linear.
  const float fscale = u_scale * (1.f / (float)D_H);
  float vr_loc[16], vi_loc[16];
#pragma unroll
  for (int i = 0; i < 16; ++i) {
    const int k  = tid + i * 256;
    const float Ur = sre[k], Ui = sim[k];
    const float Hr = heff_re[k], Hi = heff_im[k];
    vr_loc[i] = fscale * (Ur * Hr - Ui * Hi);
    vi_loc[i] = fscale * (Ur * Hi + Ui * Hr);
  }
  __syncthreads();  // all reads done before bit-reversed writes
#pragma unroll
  for (int i = 0; i < 16; ++i) {
    const int k = tid + i * 256;
    const int j = __brev((unsigned)k) >> 20;
    sre[j] = vr_loc[i];
    sim[j] = vi_loc[i];
  }
  __syncthreads();

  fft_stages(sre, sim, +1.f, tid);  // inverse FFT (1/N already folded)

  // exp map on v = Re(ifft)
  float ss2 = 0.f;
  for (int k = tid; k < D_H; k += 256) {
    const float v = sre[k];
    ss2 += v * v;
  }
  const float nv      = fmaxf(sqrtf(block_sum(ss2, red)), EPSF);
  const float arg     = sc * nv;
  const float y_scale = tanhf(arg) / arg;

  const float alpha = alpha_p[0];
  const float betap = betap_p[0];
  for (int c = tid; c < D_H; c += 256) {
    const float zv = zrow[c];
    const float h  = fmaxf(gamma1[c] * (zv - mu1[c]) * rsig1[c] + beta1[c], 0.f);
    const float y  = y_scale * sre[c];
    zr[c] = alpha * y + betap * h;  // h2, in place
  }
}

// ---------------------------------------------------------------------------
// Launch
// ---------------------------------------------------------------------------
extern "C" void kernel_launch(void* const* d_in, const int* in_sizes, int n_in,
                              void* d_out, int out_size, void* d_ws,
                              size_t ws_size, hipStream_t stream) {
  (void)in_sizes; (void)n_in; (void)out_size; (void)ws_size;

  const float* x      = (const float*)d_in[0];
  const float* W1     = (const float*)d_in[1];
  const float* b1     = (const float*)d_in[2];
  const float* gamma1 = (const float*)d_in[3];
  const float* beta1  = (const float*)d_in[4];
  const float* g_real = (const float*)d_in[5];
  const float* g_imag = (const float*)d_in[6];
  const float* alpha  = (const float*)d_in[7];
  const float* beta_p = (const float*)d_in[8];
  const float* gamma2 = (const float*)d_in[9];
  const float* beta2  = (const float*)d_in[10];
  const float* W2     = (const float*)d_in[11];
  const float* b2     = (const float*)d_in[12];
  float* out = (float*)d_out;

  float* ws      = (float*)d_ws;
  float* z1      = ws;                               // 8192*4096
  float* heff_re = z1 + (size_t)BATCH * D_H;         // 4096
  float* heff_im = heff_re + D_H;                    // 4096
  float* mu1     = heff_im + D_H;
  float* rsig1   = mu1 + D_H;
  float* mu2     = rsig1 + D_H;
  float* rsig2   = mu2 + D_H;
  float* part_s  = rsig2 + D_H;                      // 32*4096
  float* part_s2 = part_s + (size_t)32 * D_H;        // 32*4096

  heff_kernel<<<D_H / 256, 256, 0, stream>>>(g_real, g_imag, heff_re, heff_im);

  gemm1_kernel<<<dim3(D_H / TN, BATCH / TM), 256, 0, stream>>>(x, W1, b1, z1);

  bn_partial_kernel<<<dim3(D_H / 256, 32), 256, 0, stream>>>(z1, part_s, part_s2);
  bn_final_kernel<<<D_H / 256, 256, 0, stream>>>(part_s, part_s2, mu1, rsig1);

  row_kernel<<<BATCH, 256, 0, stream>>>(z1, gamma1, beta1, mu1, rsig1, heff_re,
                                        heff_im, alpha, beta_p);

  bn_partial_kernel<<<dim3(D_H / 256, 32), 256, 0, stream>>>(z1, part_s, part_s2);
  bn_final_kernel<<<D_H / 256, 256, 0, stream>>>(part_s, part_s2, mu2, rsig2);

  gemm2_kernel<<<dim3((D_OUT + TN - 1) / TN, BATCH / TM), 256, 0, stream>>>(
      z1, W2, b2, gamma2, beta2, mu2, rsig2, out);
}
